// KoLeoLoss_82557861364018
// MI455X (gfx1250) — compile-verified, matrix operand order checked by reading, not run on hardware
//
#include <hip/hip_runtime.h>
#include <math.h>

// ---------- types for WMMA ----------
typedef __bf16 v8bf  __attribute__((ext_vector_type(8)));
typedef __bf16 v16bf __attribute__((ext_vector_type(16)));
typedef float  v8f   __attribute__((ext_vector_type(8)));

#define KB_   16384   // batch
#define KD_   512     // dim
#define KTILE 128     // C tile is KTILE x KTILE
#define LDSTR 40      // padded LDS row stride in bf16 elems (80B, 16B-aligned)
#define NCHUNK 48     // 3 split-bf16 phases * (512/32)

#if __has_builtin(__builtin_amdgcn_global_load_async_to_lds_b128)
#define HAVE_ASYNC_LDS 1
#else
#define HAVE_ASYNC_LDS 0
#endif

// exact parameter types per hipcc diagnostic:
//   param1: int __attribute__((vector_size(16))) __device__ *   (addrspace(1))
//   param2: same vector type in LDS                              (addrspace(3))
typedef int v4i_t __attribute__((vector_size(16)));
typedef __attribute__((address_space(1))) v4i_t* gv4i_p;
typedef __attribute__((address_space(3))) v4i_t* lv4i_p;

__device__ __forceinline__ void wait_async_lds() {
#if HAVE_ASYNC_LDS
#if __has_builtin(__builtin_amdgcn_s_wait_asynccnt)
    __builtin_amdgcn_s_wait_asynccnt(0);
#else
    asm volatile("s_wait_asynccnt 0" ::: "memory");
#endif
#endif
}

// ---------------------------------------------------------------
// 1) split f32 -> hi/lo bf16 planes
// ---------------------------------------------------------------
__global__ __launch_bounds__(256) void koleo_prep(const float* __restrict__ x,
                                                  __bf16* __restrict__ hi,
                                                  __bf16* __restrict__ lo) {
    const int stride = gridDim.x * blockDim.x;
    for (int i = blockIdx.x * blockDim.x + threadIdx.x; i < KB_ * KD_; i += stride) {
        float v = x[i];
        __bf16 h = (__bf16)v;
        hi[i] = h;
        lo[i] = (__bf16)(v - (float)h);
    }
}

// ---------------------------------------------------------------
// 2) row norms (one wave32 per row) + init row-min bits
// ---------------------------------------------------------------
__global__ __launch_bounds__(256) void koleo_norms(const float* __restrict__ x,
                                                   float* __restrict__ sq,
                                                   int* __restrict__ rowminbits) {
    const int wave = threadIdx.x >> 5;
    const int lane = threadIdx.x & 31;
    const int row  = blockIdx.x * 8 + wave;
    const float* r = x + (size_t)row * KD_;
    float s = 0.0f;
#pragma unroll
    for (int k = 0; k < KD_ / 32; ++k) {
        float v = r[lane + k * 32];
        s = fmaf(v, v, s);
    }
#pragma unroll
    for (int off = 16; off >= 1; off >>= 1)
        s += __shfl_xor(s, off, 32);
    if (lane == 0) {
        sq[row] = s;
        rowminbits[row] = 0x7F7FFFFF;   // FLT_MAX bits (all dists are >= 0)
    }
}

// ---------------------------------------------------------------
// LDS fragment load: 16 bf16 per lane, layout per CDNA5 ISA 7.12.2
// lanes 0-15: K {kh..kh+7} U {16+kh..}, kh=0 ; lanes 16-31: kh=8
// ---------------------------------------------------------------
__device__ __forceinline__ v16bf load_frag(const __bf16* base, int row, int khalf) {
    const v8bf a = *(const v8bf*)(base + row * LDSTR + khalf);
    const v8bf b = *(const v8bf*)(base + row * LDSTR + 16 + khalf);
    return __builtin_shufflevector(a, b, 0, 1, 2, 3, 4, 5, 6, 7,
                                         8, 9, 10, 11, 12, 13, 14, 15);
}

// ---------------------------------------------------------------
// 3) main tile kernel: 128x128 distance tile per block, WMMA bf16,
//    async global->LDS staging pipelined against the matrix pipe
// ---------------------------------------------------------------
__global__ __launch_bounds__(256) void koleo_tiles(const __bf16* __restrict__ Hi,
                                                   const __bf16* __restrict__ Lo,
                                                   const float* __restrict__ sq,
                                                   int* __restrict__ rowminbits) {
    __shared__ alignas(16) __bf16 As[2][KTILE][LDSTR];
    __shared__ alignas(16) __bf16 Bs[2][KTILE][LDSTR];
    __shared__ float sqA[KTILE];
    __shared__ float sqB[KTILE];

    const int i0  = blockIdx.y * KTILE;
    const int j0  = blockIdx.x * KTILE;
    const int tid = threadIdx.x;

    if (tid < 128)      sqA[tid]       = sq[i0 + tid];
    else                sqB[tid - 128] = sq[j0 + tid - 128];

    const int wave  = tid >> 5;
    const int lane  = tid & 31;
    const int m0    = wave * 16;            // wave's row strip in the tile
    const int lrow  = lane & 15;
    const int khalf = (lane < 16) ? 0 : 8;
    const int hoff  = (lane >> 4) << 3;     // +8 rows for upper half-wave in C

    v8f acc[8] = {};

    // stage chunk c into buffer p: 128x32 bf16 A and B tiles (16B per segment)
    auto stage = [&](int c, int p) {
        const int phase = c >> 4;           // 0: hi*hi  1: hi*lo  2: lo*hi
        const int kb    = (c & 15) << 5;    // k-base in bf16 elems
        const __bf16* Asrc = (phase < 2) ? Hi : Lo;
        const __bf16* Bsrc = (phase == 1) ? Lo : Hi;
#pragma unroll
        for (int s = 0; s < 2; ++s) {
            const int seg = tid + s * 256;        // 0..511
            const int row = seg >> 2;
            const int cc  = (seg & 3) * 8;        // bf16 col within chunk
            const size_t ga = (size_t)(i0 + row) * KD_ + kb + cc;
            const size_t gb = (size_t)(j0 + row) * KD_ + kb + cc;
#if HAVE_ASYNC_LDS
            __builtin_amdgcn_global_load_async_to_lds_b128(
                (gv4i_p)(Asrc + ga), (lv4i_p)&As[p][row][cc], 0, 0);
            __builtin_amdgcn_global_load_async_to_lds_b128(
                (gv4i_p)(Bsrc + gb), (lv4i_p)&Bs[p][row][cc], 0, 0);
#else
            uint4 va = *(const uint4*)(Asrc + ga);
            uint4 vb = *(const uint4*)(Bsrc + gb);
            *(uint4*)&As[p][row][cc] = va;
            *(uint4*)&Bs[p][row][cc] = vb;
#endif
        }
    };

    // ---- software pipeline: DMA chunk c+1 while WMMA-ing chunk c ----
    stage(0, 0);
    wait_async_lds();
    __syncthreads();

    for (int c = 0; c < NCHUNK; ++c) {
        const int p = c & 1;
        if (c + 1 < NCHUNK) stage(c + 1, p ^ 1);

        const v16bf afrag = load_frag(&As[p][0][0], m0 + lrow, khalf);
#pragma unroll
        for (int n = 0; n < 8; ++n) {
            const v16bf bfrag = load_frag(&Bs[p][0][0], n * 16 + lrow, khalf);
            acc[n] = __builtin_amdgcn_wmma_f32_16x16x32_bf16(
                false, afrag, false, bfrag, (short)0, acc[n], false, false);
        }

        wait_async_lds();      // next stage's DMA done (overlapped with WMMAs)
        __syncthreads();
    }

    // ---- epilogue: distances + row-min (branchless) ----
    const float EPS = 1e-6f;
#pragma unroll
    for (int r = 0; r < 8; ++r) {
        const int rloc = m0 + r + hoff;          // row within tile
        const float sa = sqA[rloc];
        float m = 3.402823466e+38f;
#pragma unroll
        for (int n = 0; n < 8; ++n) {
            const float sb = sqB[n * 16 + lrow];
            float d2 = fmaxf(sa + sb - 2.0f * acc[n][r], 0.0f);
            float sr = sqrtf(fmaxf(d2, 1e-30f));
            float dist = (d2 > 0.0f) ? sr : 0.0f;
            m = fminf(m, dist + EPS);
        }
        // reduce over the 16 lanes holding this row's 16 columns per n-tile
#pragma unroll
        for (int off = 8; off >= 1; off >>= 1)
            m = fminf(m, __shfl_xor(m, off, 32));
        if (lrow == 0)
            atomicMin(&rowminbits[i0 + rloc], __float_as_int(m));
    }
}

// ---------------------------------------------------------------
// 4) deterministic reduction: partial sums of log(min), then combine
// ---------------------------------------------------------------
__global__ __launch_bounds__(256) void koleo_partial(const int* __restrict__ rowminbits,
                                                     float* __restrict__ partial) {
    __shared__ float red[256];
    const int tid = threadIdx.x;
    const int row = blockIdx.x * 256 + tid;
    red[tid] = logf(__int_as_float(rowminbits[row]));
    __syncthreads();
#pragma unroll
    for (int s = 128; s > 0; s >>= 1) {
        if (tid < s) red[tid] += red[tid + s];
        __syncthreads();
    }
    if (tid == 0) partial[blockIdx.x] = red[0];
}

__global__ void koleo_final(const float* __restrict__ partial, float* __restrict__ out) {
    if (threadIdx.x == 0) {
        float s = 0.0f;
        for (int i = 0; i < 64; ++i) s += partial[i];
        out[0] = -s / (float)KB_;
    }
}

// ---------------------------------------------------------------
// launcher
// ---------------------------------------------------------------
extern "C" void kernel_launch(void* const* d_in, const int* in_sizes, int n_in,
                              void* d_out, int out_size, void* d_ws, size_t ws_size,
                              hipStream_t stream) {
    (void)in_sizes; (void)n_in; (void)out_size; (void)ws_size;
    const float* feats = (const float*)d_in[0];
    char* ws = (char*)d_ws;

    const size_t plane = (size_t)KB_ * KD_ * sizeof(__bf16);   // 16 MiB
    __bf16* Hi        = (__bf16*)(ws);
    __bf16* Lo        = (__bf16*)(ws + plane);
    float*  sq        = (float*)(ws + 2 * plane);
    int*    rowmin    = (int*)(ws + 2 * plane + KB_ * sizeof(float));
    float*  partial   = (float*)(ws + 2 * plane + 2 * KB_ * sizeof(float));
    float*  out       = (float*)d_out;

    koleo_prep <<<4096, 256, 0, stream>>>(feats, Hi, Lo);
    koleo_norms<<<KB_ / 8, 256, 0, stream>>>(feats, sq, rowmin);

    dim3 grid(KB_ / KTILE, KB_ / KTILE);   // 128 x 128 tiles
    koleo_tiles<<<grid, 256, 0, stream>>>(Hi, Lo, sq, rowmin);

    koleo_partial<<<KB_ / 256, 256, 0, stream>>>(rowmin, partial);
    koleo_final  <<<1, 64, 0, stream>>>(partial, out);
}
